// LightConvEncoder_3384434230173
// MI455X (gfx1250) — matrix operand order repaired and checked by reading, not compiled, not run on hardware
//
#include <hip/hip_runtime.h>

// ---------------- types ----------------
typedef __attribute__((ext_vector_type(16))) __bf16 v16bf;
typedef __attribute__((ext_vector_type(8)))  __bf16 v8bf;
typedef __attribute__((ext_vector_type(8)))  float  v8f;

// ---------------- problem constants ----------------
constexpr int Bsz  = 8;
constexpr int Iseq = 1024;
constexpr int Dm   = 512;
constexpr int Hn   = 8;
constexpr int Kw   = 15;          // half window
constexpr int Taps = 2 * Kw + 1;  // 31
constexpr int Ln   = 6;
constexpr int Fm   = 2048;
constexpr int Mrows = Bsz * Iseq; // 8192

// ---------------- bf16 helpers (bit-level, RNE) ----------------
__device__ __forceinline__ __bf16 f2bf(float f) {
  unsigned u = __builtin_bit_cast(unsigned, f);
  u += 0x7FFFu + ((u >> 16) & 1u);
  unsigned short s = (unsigned short)(u >> 16);
  return __builtin_bit_cast(__bf16, s);
}
__device__ __forceinline__ float bflo(unsigned u) {
  return __builtin_bit_cast(float, u << 16);
}
__device__ __forceinline__ float bfhi(unsigned u) {
  return __builtin_bit_cast(float, u & 0xFFFF0000u);
}
__device__ __forceinline__ unsigned short bfbits(__bf16 b) {
  return __builtin_bit_cast(unsigned short, b);
}

// ======================================================================
// Weight transpose + fp32 -> bf16 convert:  W[K][N] -> Wt[N][K]
// ======================================================================
__global__ __launch_bounds__(256) void w_transpose_bf16(
    const float* __restrict__ in, __bf16* __restrict__ outp, int Kd, int Nd) {
  size_t idx = (size_t)blockIdx.x * 256 + threadIdx.x;
  size_t total = (size_t)Kd * Nd;
  if (idx >= total) return;
  int n = (int)(idx / (size_t)Kd);
  int k = (int)(idx % (size_t)Kd);
  outp[idx] = f2bf(in[(size_t)k * Nd + n]);
}

// ======================================================================
// Embedding + sinusoidal positional encoding -> fp32 h
// ======================================================================
__global__ __launch_bounds__(256) void embed_pe(
    const int* __restrict__ src, const float* __restrict__ E,
    float* __restrict__ h) {
  const int m = blockIdx.x;          // b*I + i
  const int i = m & (Iseq - 1);
  const int t = threadIdx.x;
  const int tok = src[m];
  const float sqrtD = 22.627416997969522f;      // sqrt(512)
  const float ln1e4 = 9.210340371976184f;       // ln(10000)
#pragma unroll
  for (int rr = 0; rr < 2; ++rr) {
    int d = t + rr * 256;
    float e = E[(size_t)tok * Dm + d] * sqrtD;
    int p = d >> 1;
    float freq = expf(-(float)(2 * p) * (ln1e4 / (float)Dm));
    float ang = (float)i * freq;
    float pe = (d & 1) ? cosf(ang) : sinf(ang);
    h[(size_t)m * Dm + d] = e + pe;
  }
}

// ======================================================================
// LayerNorm over D=512 (one row per block, 256 threads x 2 elements)
// ======================================================================
template <bool OUTBF>
__global__ __launch_bounds__(256) void lnorm_k(
    const float* __restrict__ in, const float* __restrict__ gw,
    const float* __restrict__ bw, void* __restrict__ outp) {
  __shared__ float2 red[256];
  const int row = blockIdx.x;
  const int t = threadIdx.x;
  const float* x = in + (size_t)row * Dm;
  float a = x[t];
  float b = x[t + 256];
  red[t] = make_float2(a + b, a * a + b * b);
  __syncthreads();
#pragma unroll
  for (int s = 128; s > 0; s >>= 1) {
    if (t < s) {
      red[t].x += red[t + s].x;
      red[t].y += red[t + s].y;
    }
    __syncthreads();
  }
  const float mean = red[0].x * (1.0f / Dm);
  const float var = red[0].y * (1.0f / Dm) - mean * mean;
  const float r = rsqrtf(var + 1e-5f);
  float y0 = (a - mean) * r * gw[t] + bw[t];
  float y1 = (b - mean) * r * gw[t + 256] + bw[t + 256];
  if (OUTBF) {
    __bf16* o = (__bf16*)outp;
    o[(size_t)row * Dm + t] = f2bf(y0);
    o[(size_t)row * Dm + t + 256] = f2bf(y1);
  } else {
    float* o = (float*)outp;
    o[(size_t)row * Dm + t] = y0;
    o[(size_t)row * Dm + t + 256] = y1;
  }
}

// ======================================================================
// Lightweight conv: o[b,i,h,d] = sum_k softmax(Wk[h,:])_k * v[b,i-K+k,h,d],
// masked softmax at sequence edges, then gated by g. bf16 in/out.
// ======================================================================
__global__ __launch_bounds__(256) void lightconv_k(
    const __bf16* __restrict__ v, const __bf16* __restrict__ g,
    const float* __restrict__ wk, __bf16* __restrict__ o) {
  const int m = blockIdx.x;            // b*I + i
  const int i = m & (Iseq - 1);
  const int t = threadIdx.x;
  const int d0 = t * 2;                // two channels per thread
  const int hh = d0 >> 6;              // Dh = 64
  const float* w = wk + hh * Taps;

  // masked softmax over the 31 taps
  float mx = -1e30f;
#pragma unroll
  for (int k = 0; k < Taps; ++k) {
    int j = i - Kw + k;
    if (j >= 0 && j < Iseq) mx = fmaxf(mx, w[k]);
  }
  float sum = 0.0f;
#pragma unroll
  for (int k = 0; k < Taps; ++k) {
    int j = i - Kw + k;
    if (j >= 0 && j < Iseq) sum += expf(w[k] - mx);
  }
  const float rinv = 1.0f / sum;

  float s0 = 0.0f, s1 = 0.0f;
#pragma unroll
  for (int k = 0; k < Taps; ++k) {
    int j = i - Kw + k;
    if (j >= 0 && j < Iseq) {
      float p = expf(w[k] - mx) * rinv;
      unsigned u = *(const unsigned*)&v[(size_t)(m - i + j) * Dm + d0];
      s0 += p * bflo(u);
      s1 += p * bfhi(u);
    }
  }
  unsigned gu = *(const unsigned*)&g[(size_t)m * Dm + d0];
  s0 *= bflo(gu);
  s1 *= bfhi(gu);
  unsigned ou = (unsigned)bfbits(f2bf(s0)) | ((unsigned)bfbits(f2bf(s1)) << 16);
  *(unsigned*)&o[(size_t)m * Dm + d0] = ou;
}

// ======================================================================
// WMMA GEMM:  out[M][N] = act(A[M][K] @ Bt[N][K]^T + bias) (+ resid)
// Workgroup tile 128x128 (8 wave32), 32x64 per wave, K-step 32,
// double-buffered LDS with register prefetch of the next K-slab.
// ======================================================================
template <int ACT, bool RES, bool OUTBF>   // ACT: 0 none, 1 gelu(erf), 2 relu
__global__ __launch_bounds__(256) void gemm_wmma_bf16(
    const __bf16* __restrict__ A, const __bf16* __restrict__ Bt,
    const float* __restrict__ bias, const float* __restrict__ resid,
    void* __restrict__ outp, int Nsz, int Ksz) {
  // 32 K-cols + pad to 40 (80B rows: conflict-free 16B-lane pattern)
  __shared__ __align__(16) __bf16 As[2][128][40];
  __shared__ __align__(16) __bf16 Bs[2][128][40];

  const int t = threadIdx.x;
  const int lane = t & 31;
  const int wave = t >> 5;
  const int wm = wave & 3;    // 0..3 -> 32-row slabs
  const int wn = wave >> 2;   // 0..1 -> 64-col slabs
  const int hf = lane >> 4;   // lane half (K-offset select)
  const int lrow = lane & 15;
  const int m0 = blockIdx.y * 128;
  const int n0 = blockIdx.x * 128;

  v8f acc[2][4];
#pragma unroll
  for (int mi = 0; mi < 2; ++mi)
#pragma unroll
    for (int ni = 0; ni < 4; ++ni)
#pragma unroll
      for (int j = 0; j < 8; ++j) acc[mi][ni][j] = 0.0f;

  const int srow = t >> 2;          // 0..63
  const int scol = (t & 3) * 8;     // 0,8,16,24 (bf16 elems)
  const __bf16* Arow0 = A + (size_t)(m0 + srow) * Ksz + scol;
  const __bf16* Arow1 = A + (size_t)(m0 + srow + 64) * Ksz + scol;
  const __bf16* Brow0 = Bt + (size_t)(n0 + srow) * Ksz + scol;
  const __bf16* Brow1 = Bt + (size_t)(n0 + srow + 64) * Ksz + scol;

  uint4 pa0, pa1, pb0, pb1;
  pa0 = *(const uint4*)(Arow0);
  pa1 = *(const uint4*)(Arow1);
  pb0 = *(const uint4*)(Brow0);
  pb1 = *(const uint4*)(Brow1);
  *(uint4*)(&As[0][srow][scol]) = pa0;
  *(uint4*)(&As[0][srow + 64][scol]) = pa1;
  *(uint4*)(&Bs[0][srow][scol]) = pb0;
  *(uint4*)(&Bs[0][srow + 64][scol]) = pb1;
  __syncthreads();

  const int nk = Ksz >> 5;
  for (int kt = 0; kt < nk; ++kt) {
    const int cur = kt & 1;
    const bool more = (kt + 1) < nk;
    if (more) {
      const int k1 = (kt + 1) * 32;
      pa0 = *(const uint4*)(Arow0 + k1);
      pa1 = *(const uint4*)(Arow1 + k1);
      pb0 = *(const uint4*)(Brow0 + k1);
      pb1 = *(const uint4*)(Brow1 + k1);
    }

    // A fragment: 16x32 bf16. lanes 0-15: K 0..7 / 16..23; lanes 16-31: +8.
    v16bf af[2];
#pragma unroll
    for (int mi = 0; mi < 2; ++mi) {
      const __bf16* p = &As[cur][wm * 32 + mi * 16 + lrow][hf * 8];
      v8bf lo = *(const v8bf*)(p);
      v8bf hi = *(const v8bf*)(p + 16);
      af[mi] = __builtin_shufflevector(lo, hi, 0, 1, 2, 3, 4, 5, 6, 7, 8, 9,
                                       10, 11, 12, 13, 14, 15);
    }
    // B fragment: 32x16 bf16. lanes 0-15: K 0..7 / 16..23; lanes 16-31: +8.
    v16bf bfr[4];
#pragma unroll
    for (int ni = 0; ni < 4; ++ni) {
      const __bf16* p = &Bs[cur][wn * 64 + ni * 16 + lrow][hf * 8];
      v8bf lo = *(const v8bf*)(p);
      v8bf hi = *(const v8bf*)(p + 16);
      bfr[ni] = __builtin_shufflevector(lo, hi, 0, 1, 2, 3, 4, 5, 6, 7, 8, 9,
                                        10, 11, 12, 13, 14, 15);
    }
#pragma unroll
    for (int mi = 0; mi < 2; ++mi)
#pragma unroll
      for (int ni = 0; ni < 4; ++ni)
        acc[mi][ni] = __builtin_amdgcn_wmma_f32_16x16x32_bf16(
            false, af[mi], false, bfr[ni], (short)0, acc[mi][ni], false, false);

    if (more) {
      __syncthreads();  // all waves done reading buf[cur^1] two steps ago
      const int nxt = cur ^ 1;
      *(uint4*)(&As[nxt][srow][scol]) = pa0;
      *(uint4*)(&As[nxt][srow + 64][scol]) = pa1;
      *(uint4*)(&Bs[nxt][srow][scol]) = pb0;
      *(uint4*)(&Bs[nxt][srow + 64][scol]) = pb1;
      __syncthreads();
    }
  }

  // epilogue: C/D layout: VGPR r -> M = r + 8*hf, N = lrow
#pragma unroll
  for (int mi = 0; mi < 2; ++mi) {
#pragma unroll
    for (int ni = 0; ni < 4; ++ni) {
      const int mb = m0 + wm * 32 + mi * 16 + hf * 8;
      const int n = n0 + wn * 64 + ni * 16 + lrow;
      const float bn = bias[n];
#pragma unroll
      for (int r = 0; r < 8; ++r) {
        float v = acc[mi][ni][r] + bn;
        if (ACT == 1) v = 0.5f * v * (1.0f + erff(v * 0.70710678118654752f));
        if (ACT == 2) v = fmaxf(v, 0.0f);
        size_t idx = (size_t)(mb + r) * Nsz + n;
        if (RES) v += resid[idx];
        if (OUTBF)
          ((__bf16*)outp)[idx] = f2bf(v);
        else
          ((float*)outp)[idx] = v;
      }
    }
  }
}

// ======================================================================
// Host side
// ======================================================================
extern "C" void kernel_launch(void* const* d_in, const int* in_sizes, int n_in,
                              void* d_out, int out_size, void* d_ws,
                              size_t ws_size, hipStream_t stream) {
  // input order: src, E, ln1_g, ln1_b, Wv, bv, Wg, bg, lnv_g, lnv_b, Wk,
  //              Wo, bo, ln2_g, ln2_b, W1, b1, W2, b2, lnf_g, lnf_b
  const int*   src   = (const int*)d_in[0];
  const float* E     = (const float*)d_in[1];
  const float* ln1_g = (const float*)d_in[2];
  const float* ln1_b = (const float*)d_in[3];
  const float* Wv    = (const float*)d_in[4];
  const float* bv    = (const float*)d_in[5];
  const float* Wg    = (const float*)d_in[6];
  const float* bg    = (const float*)d_in[7];
  const float* lnv_g = (const float*)d_in[8];
  const float* lnv_b = (const float*)d_in[9];
  const float* Wk    = (const float*)d_in[10];
  const float* Wo    = (const float*)d_in[11];
  const float* bo    = (const float*)d_in[12];
  const float* ln2_g = (const float*)d_in[13];
  const float* ln2_b = (const float*)d_in[14];
  const float* W1    = (const float*)d_in[15];
  const float* b1    = (const float*)d_in[16];
  const float* W2    = (const float*)d_in[17];
  const float* b2    = (const float*)d_in[18];
  const float* lnf_g = (const float*)d_in[19];
  const float* lnf_b = (const float*)d_in[20];

  // workspace carve-up (~135 MB)
  char* ws = (char*)d_ws;
  size_t off = 0;
  auto alloc = [&](size_t bytes) -> void* {
    void* p = ws + off;
    off += (bytes + 255) & ~(size_t)255;
    return p;
  };
  float*  h    = (float*)alloc((size_t)Mrows * Dm * 4);
  float*  vtmp = (float*)alloc((size_t)Mrows * Dm * 4);
  __bf16* xb   = (__bf16*)alloc((size_t)Mrows * Dm * 2);
  __bf16* vb   = (__bf16*)alloc((size_t)Mrows * Dm * 2);
  __bf16* gb   = (__bf16*)alloc((size_t)Mrows * Dm * 2);
  __bf16* ob   = (__bf16*)alloc((size_t)Mrows * Dm * 2);
  __bf16* tb   = (__bf16*)alloc((size_t)Mrows * Fm * 2);
  const size_t perLayerW =
      (size_t)3 * Dm * Dm + (size_t)2 * Dm * Fm;  // bf16 elems per layer
  __bf16* Wt = (__bf16*)alloc(perLayerW * Ln * 2);

  // one-time weight transpose+convert (fp32 [K][N] -> bf16 [N][K])
  for (int l = 0; l < Ln; ++l) {
    __bf16* wvt = Wt + perLayerW * l;
    __bf16* wgt = wvt + (size_t)Dm * Dm;
    __bf16* wot = wgt + (size_t)Dm * Dm;
    __bf16* w1t = wot + (size_t)Dm * Dm;
    __bf16* w2t = w1t + (size_t)Dm * Fm;
    int ndd = (Dm * Dm + 255) / 256;
    int ndf = (Dm * Fm + 255) / 256;
    w_transpose_bf16<<<ndd, 256, 0, stream>>>(Wv + (size_t)l * Dm * Dm, wvt, Dm, Dm);
    w_transpose_bf16<<<ndd, 256, 0, stream>>>(Wg + (size_t)l * Dm * Dm, wgt, Dm, Dm);
    w_transpose_bf16<<<ndd, 256, 0, stream>>>(Wo + (size_t)l * Dm * Dm, wot, Dm, Dm);
    w_transpose_bf16<<<ndf, 256, 0, stream>>>(W1 + (size_t)l * Dm * Fm, w1t, Dm, Fm);
    w_transpose_bf16<<<ndf, 256, 0, stream>>>(W2 + (size_t)l * Fm * Dm, w2t, Fm, Dm);
  }

  embed_pe<<<Mrows, 256, 0, stream>>>(src, E, h);

  const dim3 gD(Dm / 128, Mrows / 128);   // N=512 GEMMs: 4 x 64 blocks
  const dim3 gF(Fm / 128, Mrows / 128);   // N=2048 GEMM: 16 x 64 blocks

  for (int l = 0; l < Ln; ++l) {
    __bf16* wvt = Wt + perLayerW * l;
    __bf16* wgt = wvt + (size_t)Dm * Dm;
    __bf16* wot = wgt + (size_t)Dm * Dm;
    __bf16* w1t = wot + (size_t)Dm * Dm;
    __bf16* w2t = w1t + (size_t)Dm * Fm;

    lnorm_k<true><<<Mrows, 256, 0, stream>>>(h, ln1_g + l * Dm, ln1_b + l * Dm, xb);
    gemm_wmma_bf16<1, false, false><<<gD, 256, 0, stream>>>(
        xb, wvt, bv + l * Dm, nullptr, vtmp, Dm, Dm);
    gemm_wmma_bf16<1, false, true><<<gD, 256, 0, stream>>>(
        xb, wgt, bg + l * Dm, nullptr, gb, Dm, Dm);
    lnorm_k<true><<<Mrows, 256, 0, stream>>>(vtmp, lnv_g + l * Dm, lnv_b + l * Dm, vb);
    lightconv_k<<<Mrows, 256, 0, stream>>>(vb, gb, Wk + (size_t)l * Hn * Taps, ob);
    gemm_wmma_bf16<0, true, false><<<gD, 256, 0, stream>>>(
        ob, wot, bo + l * Dm, h, h, Dm, Dm);
    lnorm_k<true><<<Mrows, 256, 0, stream>>>(h, ln2_g + l * Dm, ln2_b + l * Dm, xb);
    gemm_wmma_bf16<2, false, true><<<gF, 256, 0, stream>>>(
        xb, w1t, b1 + (size_t)l * Fm, nullptr, tb, Fm, Dm);
    gemm_wmma_bf16<0, true, false><<<gD, 256, 0, stream>>>(
        tb, w2t, b2 + l * Dm, h, h, Dm, Fm);
  }

  lnorm_k<false><<<Mrows, 256, 0, stream>>>(h, lnf_g, lnf_b, d_out);
}
